// HyperFlowNet_50792283242752
// MI455X (gfx1250) — matrix operand order repaired and checked by reading, not compiled, not run on hardware
//
#include <hip/hip_runtime.h>
#include <hip/hip_bf16.h>

typedef __attribute__((ext_vector_type(16))) __bf16 v16bf;
typedef __attribute__((ext_vector_type(8)))  __bf16 v8bf;
typedef __attribute__((ext_vector_type(8)))  float  v8f;

constexpr int BB = 2, NN = 50000, EE = 400000, CIN = 5, CC = 128, SS = 32, LL = 4, FF = 512;
constexpr float BETA_ = 1.0f;
constexpr float PIF = 3.14159265358979323846f;

// ---------------- helpers ----------------

__device__ inline void atomAdd(float* p, float v) {
  (void)__hip_atomic_fetch_add(p, v, __ATOMIC_RELAXED, __HIP_MEMORY_SCOPE_AGENT);
}

// Fragments live pre-swizzled (both in LDS for A and in global for B):
// tile t, lane l own 16 contiguous bf16 at [(t*32 + l)*16]. One 32B vector load.
__device__ inline v16bf frag_swz(const __bf16* base, int tile, int lane) {
  return ((const v16bf*)base)[tile * 32 + lane];
}

__device__ inline v8f wmma_bf(v16bf a, v16bf b, v8f c) {
  return __builtin_amdgcn_wmma_f32_16x16x32_bf16(false, a, false, b, (short)0, c, false, false);
}

// A-fragment swizzle (16x32 bf16 tile, ISA table):
//  lane<16: m=lane,   e0..7 -> K 0..7,   e8..15 -> K 16..23
//  lane>=16: m=lane-16, e0..7 -> K 8..15, e8..15 -> K 24..31
// inverse for an 8-wide K-chunk kb in {0,8,16,24} of row m:
//  lane = m + ((kb&8)?16:0), elem base = (kb&16)?8:0
__device__ inline int a_swz_idx(int rt, int ktiles, int kt, int m, int kb) {
  int lane = m + ((kb & 8) ? 16 : 0);
  int eb = (kb & 16) ? 8 : 0;
  return ((rt * ktiles + kt) * 32 + lane) * 16 + eb;
}

// ---------------- elementwise kernels ----------------

__global__ void k_zero(float* __restrict__ p, long long n) {
  long long i = (long long)blockIdx.x * blockDim.x + threadIdx.x;
  if (i < n) p[i] = 0.f;
}

// Convert W (K x Nout, row-major f32) into swizzled bf16 B-fragments:
// dst[((kt*(Nout/16)+nt)*32 + lane)*16 + e] = W[(kt*32 + ((lane>=16)?8:0) + (e&7) + ((e&8)?16:0))*Nout + nt*16 + (lane&15)]
__global__ void k_cvt_swz(const float* __restrict__ W, __bf16* __restrict__ dst,
                          int K, int Nout, float scale) {
  long long i = (long long)blockIdx.x * blockDim.x + threadIdx.x;
  if (i >= (long long)K * Nout) return;
  int e = (int)(i & 15);
  int lane = (int)((i >> 4) & 31);
  long long tile = i >> 9;
  int ntn = Nout >> 4;
  int kt = (int)(tile / ntn);
  int nt = (int)(tile % ntn);
  int k = kt * 32 + ((lane & 16) ? 8 : 0) + (e & 7) + ((e & 8) ? 16 : 0);
  int n = nt * 16 + (lane & 15);
  dst[i] = (__bf16)(W[(size_t)k * Nout + n] * scale);
}

__global__ void k_subcvt(const float* __restrict__ h, const float* __restrict__ hl,
                         __bf16* __restrict__ d, long long n) {
  long long i = (long long)blockIdx.x * blockDim.x + threadIdx.x;
  if (i < n) d[i] = (__bf16)(h[i] - hl[i]);
}

// ---------------- embed ----------------

__global__ __launch_bounds__(128) void k_embed(const float* __restrict__ inp,
    const float* __restrict__ coords, const float* __restrict__ tnorm,
    const float* __restrict__ freq, const float* __restrict__ W,
    const float* __restrict__ bias, float* __restrict__ x) {
  int node = blockIdx.x;                 // b*NN + n
  int b = node / NN;
  __shared__ float feat[32];
  int t = threadIdx.x;
  if (t < 29) {
    float v;
    if (t < 5) {
      v = inp[(size_t)node * CIN + t];
    } else if (t < 21) {
      int j = (t - 5) & 7;
      float cx = coords[(size_t)node * 2 + 0], cy = coords[(size_t)node * 2 + 1];
      float p = 2.f * PIF * (cx * freq[j] + cy * freq[8 + j]);
      v = (t < 13) ? sinf(p) : cosf(p);
    } else {
      int i = (t - 21) & 3;
      float omega = powf(1000.f, -(float)i / 4.f);
      float ang = omega * tnorm[b] * 1000.f;
      v = (t < 25) ? sinf(ang) : cosf(ang);
    }
    feat[t] = v;
  }
  __syncthreads();
  float acc = bias[t];
#pragma unroll
  for (int k = 0; k < 29; ++k) acc += feat[k] * W[k * CC + t];
  x[(size_t)node * CC + t] = acc;
}

// ---------------- layernorm (block = 128 threads = 1 node) ----------------

__global__ __launch_bounds__(128) void k_ln(const float* __restrict__ in,
    const float* __restrict__ g, const float* __restrict__ bvec,
    float* __restrict__ out32, __bf16* __restrict__ outbf) {
  int node = blockIdx.x;
  int c = threadIdx.x;
  __shared__ float red[128];
  float v = in[(size_t)node * CC + c];
  red[c] = v;
  __syncthreads();
  for (int s = 64; s > 0; s >>= 1) { if (c < s) red[c] += red[c + s]; __syncthreads(); }
  float mean = red[0] * (1.f / CC);
  __syncthreads();
  float d = v - mean;
  red[c] = d * d;
  __syncthreads();
  for (int s = 64; s > 0; s >>= 1) { if (c < s) red[c] += red[c + s]; __syncthreads(); }
  float var = red[0] * (1.f / CC);
  float y = d * rsqrtf(var + 1e-5f) * g[c] + bvec[c];
  if (out32) out32[(size_t)node * CC + c] = y;
  if (outbf) outbf[(size_t)node * CC + c] = (__bf16)y;
}

// ---------------- sparse aggregation (edge-parallel fp32 atomics) ----------------

__global__ void k_spmm(const int* __restrict__ rows, const int* __restrict__ cols,
                       const float* __restrict__ vals, const float* __restrict__ h,
                       float* __restrict__ hloc) {
  long long idx = (long long)blockIdx.x * blockDim.x + threadIdx.x;
  long long tot = (long long)BB * EE * 4;
  if (idx >= tot) return;
  int q = (int)(idx & 3);
  long long be = idx >> 2;
  int e = (int)(be % EE);
  int b = (int)(be / EE);
  int r  = rows[e];
  int cl = cols[e];
  float a = vals[e];
  const float* src = h    + ((size_t)b * NN + cl) * CC + q * 32;
  float*       dst = hloc + ((size_t)b * NN + r ) * CC + q * 32;
#pragma unroll
  for (int i = 0; i < 8; ++i) {
    float4 v = ((const float4*)src)[i];
    atomAdd(dst + 4 * i + 0, a * v.x);
    atomAdd(dst + 4 * i + 1, a * v.y);
    atomAdd(dst + 4 * i + 2, a * v.z);
    atomAdd(dst + 4 * i + 3, a * v.w);
  }
}

// ---------------- fused logits + softmax + wsum + slices-partial ----------------
// grid (ceil(N/128), B), 256 threads (8 waves). A tiles staged swizzled in LDS.

__global__ __launch_bounds__(256) void k_logits(const __bf16* __restrict__ hbf,
    const __bf16* __restrict__ hfbf, const __bf16* __restrict__ WsB,
    const __bf16* __restrict__ WfB, const float* __restrict__ bsl,
    const float* __restrict__ bfr, float* __restrict__ wout,
    __bf16* __restrict__ wbf, float* __restrict__ wsum, float* __restrict__ slices) {
  __shared__ alignas(32) __bf16 tAsw[8 * 4 * 32 * 16];   // 8 row-tiles x 4 k-tiles, 32KB
  __shared__ float lg[128][32];                          // logits -> softmax weights
  int b    = blockIdx.y;
  int base = blockIdx.x * 128;
  int tid = threadIdx.x, lane = tid & 31, wv = tid >> 5;
  const __bf16* hp  = hbf  + (size_t)b * NN * CC;
  const __bf16* hfp = hfbf + (size_t)b * NN * CC;

  // stage h_front tile (swizzled, 16B chunks)
  for (int i = tid; i < 128 * 16; i += 256) {
    int rr = i >> 4, ck = (i & 15) << 3;
    int n = base + rr;
    v8bf v = {};
    if (n < NN) v = *(const v8bf*)(hfp + (size_t)n * CC + ck);
    *(v8bf*)(tAsw + a_swz_idx(rr >> 4, 4, ck >> 5, rr & 15, ck & 31)) = v;
  }
  // prefetch the h tile we will stage next, so it lands in cache during WMMA pass 1
  {
    int rr = tid >> 1, ck = (tid & 1) * 64;
    int n = base + rr;
    if (n < NN) __builtin_prefetch(hp + (size_t)n * CC + ck, 0, 1);
  }
  __syncthreads();
  v8f acc0 = {}, acc1 = {};
#pragma unroll
  for (int ks = 0; ks < 4; ++ks) {
    v16bf a = frag_swz(tAsw, wv * 4 + ks, lane);
    acc0 = wmma_bf(a, frag_swz(WfB, ks * 2 + 0, lane), acc0);
    acc1 = wmma_bf(a, frag_swz(WfB, ks * 2 + 1, lane), acc1);
  }
  __syncthreads();
  // stage h tile (stays resident for the slices reduction below)
  for (int i = tid; i < 128 * 16; i += 256) {
    int rr = i >> 4, ck = (i & 15) << 3;
    int n = base + rr;
    v8bf v = {};
    if (n < NN) v = *(const v8bf*)(hp + (size_t)n * CC + ck);
    *(v8bf*)(tAsw + a_swz_idx(rr >> 4, 4, ck >> 5, rr & 15, ck & 31)) = v;
  }
  __syncthreads();
#pragma unroll
  for (int ks = 0; ks < 4; ++ks) {
    v16bf a = frag_swz(tAsw, wv * 4 + ks, lane);
    acc0 = wmma_bf(a, frag_swz(WsB, ks * 2 + 0, lane), acc0);
    acc1 = wmma_bf(a, frag_swz(WsB, ks * 2 + 1, lane), acc1);
  }
  {
    int n16 = lane & 15, rb = (lane & 16) ? 8 : 0;
#pragma unroll
    for (int r = 0; r < 8; ++r) {
      int row = wv * 16 + rb + r;
      lg[row][n16]      = acc0[r] + bsl[n16]      + BETA_ * bfr[n16];
      lg[row][16 + n16] = acc1[r] + bsl[16 + n16] + BETA_ * bfr[16 + n16];
    }
  }
  __syncthreads();
  // softmax per node row (threads 0..127)
  if (tid < 128) {
    int n = base + tid;
    if (n < NN) {
      float vv[SS];
      float mx = -3.4e38f;
#pragma unroll
      for (int s = 0; s < SS; ++s) { vv[s] = lg[tid][s]; mx = fmaxf(mx, vv[s]); }
      float sum = 0.f;
#pragma unroll
      for (int s = 0; s < SS; ++s) { vv[s] = expf(vv[s] - mx); sum += vv[s]; }
      float inv = 1.f / sum;
      size_t wb = ((size_t)b * NN + n) * SS;
#pragma unroll
      for (int s = 0; s < SS; ++s) {
        float w = vv[s] * inv;
        lg[tid][s] = w;
        wout[wb + s] = w;
        wbf[wb + s]  = (__bf16)w;
      }
    } else {
#pragma unroll
      for (int s = 0; s < SS; ++s) lg[tid][s] = 0.f;   // zero so reductions ignore pad rows
    }
  }
  __syncthreads();
  // block-partial wsum (one atomic per (b,s) per block)
  if (tid < SS) {
    float s = 0.f;
    for (int r = 0; r < 128; ++r) s += lg[r][tid];
    atomAdd(&wsum[b * SS + tid], s);
  }
  // block-partial slices[s][c]: thread -> s = tid>>3, c = (tid&7)*16..+15
  {
    int s = tid >> 3, cb = (tid & 7) * 16;
    float acc[16];
#pragma unroll
    for (int i = 0; i < 16; ++i) acc[i] = 0.f;
    for (int r = 0; r < 128; ++r) {
      float w = lg[r][s];
      v8bf a0 = *(const v8bf*)(tAsw + a_swz_idx(r >> 4, 4, cb >> 5, r & 15, cb & 31));
      v8bf a1 = *(const v8bf*)(tAsw + a_swz_idx(r >> 4, 4, (cb + 8) >> 5, r & 15, (cb + 8) & 31));
#pragma unroll
      for (int i = 0; i < 8; ++i) {
        acc[i]     += w * (float)a0[i];
        acc[8 + i] += w * (float)a1[i];
      }
    }
    float* sp = slices + ((size_t)b * SS + s) * CC + cb;
#pragma unroll
    for (int i = 0; i < 16; ++i) atomAdd(sp + i, acc[i]);
  }
}

// ---------------- slice-token MHA (32 tokens, tiny: scalar) ----------------

__global__ __launch_bounds__(256) void k_mha(const float* __restrict__ slices,
    const float* __restrict__ wsum, const float* __restrict__ Win,
    const float* __restrict__ bin, const float* __restrict__ Wout,
    const float* __restrict__ bout, __bf16* __restrict__ sobf) {
  int b = blockIdx.x, tid = threadIdx.x;
  __shared__ float sl[32][128];
  __shared__ float qS[32][128], kS[32][128], vS[32][128];
  for (int i = tid; i < SS * CC; i += 256) {
    int s = i >> 7, c = i & 127;
    float ws = fmaxf(wsum[b * SS + s], 1e-8f);
    sl[s][c] = slices[((size_t)b * SS + s) * CC + c] / ws;
  }
  __syncthreads();
  for (int i = tid; i < SS * 3 * CC; i += 256) {
    int t = i / (3 * CC), j = i % (3 * CC);
    float a = bin[j];
    for (int c = 0; c < CC; ++c) a += sl[t][c] * Win[c * (3 * CC) + j];
    if (j < CC) qS[t][j] = a;
    else if (j < 2 * CC) kS[t][j - CC] = a;
    else vS[t][j - 2 * CC] = a;
  }
  __syncthreads();
  {
    int hh = tid >> 5, qi = tid & 31;   // 8 heads x 32 queries
    float sc[32];
    float mx = -3.4e38f;
#pragma unroll
    for (int j = 0; j < 32; ++j) {
      float d = 0.f;
#pragma unroll
      for (int dd = 0; dd < 16; ++dd) d += qS[qi][hh * 16 + dd] * kS[j][hh * 16 + dd];
      sc[j] = d * 0.25f;                // 1/sqrt(HD=16)
      mx = fmaxf(mx, sc[j]);
    }
    float sum = 0.f;
#pragma unroll
    for (int j = 0; j < 32; ++j) { sc[j] = expf(sc[j] - mx); sum += sc[j]; }
    float inv = 1.f / sum;
    float o[16];
#pragma unroll
    for (int dd = 0; dd < 16; ++dd) o[dd] = 0.f;
#pragma unroll
    for (int j = 0; j < 32; ++j) {
      float w = sc[j] * inv;
#pragma unroll
      for (int dd = 0; dd < 16; ++dd) o[dd] += w * vS[j][hh * 16 + dd];
    }
#pragma unroll
    for (int dd = 0; dd < 16; ++dd) sl[qi][hh * 16 + dd] = o[dd];  // sl now holds o
  }
  __syncthreads();
  // slices_out written directly in swizzled B-fragment layout (K=32, Nout=128)
  for (int i = tid; i < SS * CC; i += 256) {
    int t = i >> 7, c = i & 127;               // t = K index, c = N index
    float a = bout[c];
    for (int k = 0; k < CC; ++k) a += sl[t][k] * Wout[k * CC + c];
    int idx = (((c >> 4) * 32) + ((t & 8) ? 16 : 0) + (c & 15)) * 16 + (t & 7) + ((t & 16) >> 1);
    sobf[(size_t)b * SS * CC + idx] = (__bf16)a;
  }
}

// ---------------- x += w (N x 32) @ slices_out (32 x 128) via WMMA ----------------

__global__ __launch_bounds__(256) void k_xupd(const __bf16* __restrict__ wbf,
    const __bf16* __restrict__ so, float* __restrict__ x) {
  __shared__ alignas(32) __bf16 aT[8 * 32 * 16];   // 8 row-tiles x 1 k-tile, 8KB
  int b = blockIdx.y;
  int base = blockIdx.x * 128;
  int tid = threadIdx.x, lane = tid & 31, wv = tid >> 5;
  for (int i = tid; i < 128 * 4; i += 256) {
    int rr = i >> 2, ck = (i & 3) << 3;
    int n = base + rr;
    v8bf v = {};
    if (n < NN) v = *(const v8bf*)(wbf + ((size_t)b * NN + n) * SS + ck);
    *(v8bf*)(aT + a_swz_idx(rr >> 4, 1, 0, rr & 15, ck & 31)) = v;
  }
  __syncthreads();
  v16bf a = frag_swz(aT, wv, lane);
  const __bf16* Bp = so + (size_t)b * SS * CC;
  int n16 = lane & 15, rb = (lane & 16) ? 8 : 0;
#pragma unroll
  for (int ct = 0; ct < 8; ++ct) {
    v8f acc = {};
    acc = wmma_bf(a, frag_swz(Bp, ct, lane), acc);
#pragma unroll
    for (int r = 0; r < 8; ++r) {
      int n = base + wv * 16 + rb + r;
      if (n < NN) {
        size_t xi = ((size_t)b * NN + n) * CC + ct * 16 + n16;
        x[xi] += acc[r];
      }
    }
  }
}

// ---------------- fused FFN: x += gelu(h2@W1 + b1) @ W2 + b2 ----------------
// grid (ceil(N/32), B), 256 threads. Gelu intermediate (32x512 bf16) lives in LDS.

__global__ __launch_bounds__(256) void k_ffn(const __bf16* __restrict__ h2,
    const __bf16* __restrict__ W1, const float* __restrict__ b1,
    const __bf16* __restrict__ W2, const float* __restrict__ b2,
    float* __restrict__ x) {
  __shared__ alignas(32) __bf16 tA[2 * 4 * 32 * 16];    // 32x128 swizzled, 8KB
  __shared__ alignas(32) __bf16 G[2 * 16 * 32 * 16];    // 32x512 swizzled, 32KB
  int b = blockIdx.y;
  int base = blockIdx.x * 32;
  int tid = threadIdx.x, lane = tid & 31, wv = tid >> 5;
  for (int i = tid; i < 32 * 16; i += 256) {
    int rr = i >> 4, ck = (i & 15) << 3;
    int n = base + rr;
    v8bf v = {};
    if (n < NN) v = *(const v8bf*)(h2 + ((size_t)b * NN + n) * CC + ck);
    *(v8bf*)(tA + a_swz_idx(rr >> 4, 4, ck >> 5, rr & 15, ck & 31)) = v;
  }
  __syncthreads();
  int n16 = lane & 15, rb = (lane & 16) ? 8 : 0;
  // GEMM1: (32x128)x(128x512): 64 wave-tiles, 8 per wave
  for (int it = 0; it < 8; ++it) {
    int t = wv * 8 + it;
    int rt = t & 1, ct = t >> 1;
    v8f acc = {};
#pragma unroll
    for (int ks = 0; ks < 4; ++ks) {
      v16bf a = frag_swz(tA, rt * 4 + ks, lane);
      acc = wmma_bf(a, frag_swz(W1, ks * 32 + ct, lane), acc);
    }
#pragma unroll
    for (int r = 0; r < 8; ++r) {
      int row = rt * 16 + rb + r;
      int col = ct * 16 + n16;
      float v = acc[r] + b1[col];
      float g = 0.5f * v * (1.f + erff(v * 0.70710678118f));   // exact gelu
      int kk = col & 31;
      int gi = (((row >> 4) * 16 + (col >> 5)) * 32 + (row & 15) + ((kk & 8) ? 16 : 0)) * 16
               + ((kk & 16) ? 8 : 0) + (kk & 7);
      G[gi] = (__bf16)g;
    }
  }
  __syncthreads();
  // GEMM2: (32x512)x(512x128): 16 wave-tiles, 2 per wave, K=512
  for (int it = 0; it < 2; ++it) {
    int t = wv * 2 + it;
    int rt = t & 1, ct = t >> 1;
    v8f acc = {};
#pragma unroll
    for (int ks = 0; ks < 16; ++ks) {
      v16bf a = frag_swz(G, rt * 16 + ks, lane);
      acc = wmma_bf(a, frag_swz(W2, ks * 8 + ct, lane), acc);
    }
#pragma unroll
    for (int r = 0; r < 8; ++r) {
      int n = base + rt * 16 + rb + r;
      if (n < NN) {
        int col = ct * 16 + n16;
        size_t xi = ((size_t)b * NN + n) * CC + col;
        x[xi] += acc[r] + b2[col];
      }
    }
  }
}

// ---------------- output projection ----------------

__global__ void k_proj(const float* __restrict__ x, const float* __restrict__ W,
                       const float* __restrict__ bvec, float* __restrict__ out) {
  long long idx = (long long)blockIdx.x * blockDim.x + threadIdx.x;
  if (idx >= (long long)BB * NN) return;
  float acc[CIN];
#pragma unroll
  for (int o = 0; o < CIN; ++o) acc[o] = bvec[o];
  const float* xp = x + (size_t)idx * CC;
  for (int c = 0; c < CC; ++c) {
    float xv = xp[c];
#pragma unroll
    for (int o = 0; o < CIN; ++o) acc[o] += xv * W[c * CIN + o];
  }
#pragma unroll
  for (int o = 0; o < CIN; ++o) out[(size_t)idx * CIN + o] = acc[o];
}

// ---------------- host driver ----------------

extern "C" void kernel_launch(void* const* d_in, const int* in_sizes, int n_in,
                              void* d_out, int out_size, void* d_ws, size_t ws_size,
                              hipStream_t stream) {
  (void)in_sizes; (void)n_in; (void)out_size; (void)ws_size;
  const float* inputs  = (const float*)d_in[0];
  const float* coords  = (const float*)d_in[1];
  const float* t_norm  = (const float*)d_in[2];
  const int*   adj_idx = (const int*)d_in[3];
  const float* adj_val = (const float*)d_in[4];
  const float* freq    = (const float*)d_in[5];
  const float* embed_W = (const float*)d_in[6];
  const float* embed_b = (const float*)d_in[7];
  const float* ln1_g   = (const float*)d_in[8];
  const float* ln1_b   = (const float*)d_in[9];
  const float* Wslice  = (const float*)d_in[10];
  const float* bslice  = (const float*)d_in[11];
  const float* Wfront  = (const float*)d_in[12];
  const float* bfront  = (const float*)d_in[13];
  const float* Win     = (const float*)d_in[14];
  const float* b_in    = (const float*)d_in[15];
  const float* Wout    = (const float*)d_in[16];
  const float* bout    = (const float*)d_in[17];
  const float* ln2_g   = (const float*)d_in[18];
  const float* ln2_b   = (const float*)d_in[19];
  const float* Wff1    = (const float*)d_in[20];
  const float* bff1    = (const float*)d_in[21];
  const float* Wff2    = (const float*)d_in[22];
  const float* bff2    = (const float*)d_in[23];
  const float* proj_W  = (const float*)d_in[24];
  const float* proj_b  = (const float*)d_in[25];
  float* outp = (float*)d_out;

  char* ws = (char*)d_ws;
  size_t off = 0;
  auto alloc = [&](size_t bytes) -> void* {
    void* p = ws + off;
    off = (off + bytes + 255) & ~(size_t)255;
    return p;
  };
  float*  x      = (float*) alloc((size_t)BB * NN * CC * 4);
  float*  h      = (float*) alloc((size_t)BB * NN * CC * 4);
  float*  hloc   = (float*) alloc((size_t)BB * NN * CC * 4);
  __bf16* hbf    = (__bf16*)alloc((size_t)BB * NN * CC * 2);
  __bf16* hfbf   = (__bf16*)alloc((size_t)BB * NN * CC * 2);   // also reused as h2 bf16
  __bf16* wbf    = (__bf16*)alloc((size_t)BB * NN * SS * 2);
  float*  wsum   = (float*) alloc((size_t)BB * SS * 4);
  float*  slices = (float*) alloc((size_t)BB * SS * CC * 4);
  __bf16* sobf   = (__bf16*)alloc((size_t)BB * SS * CC * 2);
  __bf16* WslB   = (__bf16*)alloc((size_t)LL * CC * SS * 2);
  __bf16* WfrB   = (__bf16*)alloc((size_t)LL * CC * SS * 2);
  __bf16* Wf1B   = (__bf16*)alloc((size_t)LL * CC * FF * 2);
  __bf16* Wf2B   = (__bf16*)alloc((size_t)LL * FF * CC * 2);

  // per-layer weight conversion into swizzled WMMA fragment layout
  auto cvt_swz = [&](const float* s, __bf16* d, int K, int Nout, float scale) {
    long long cnt = (long long)K * Nout;
    k_cvt_swz<<<dim3((unsigned)((cnt + 255) / 256)), dim3(256), 0, stream>>>(s, d, K, Nout, scale);
  };
  for (int l = 0; l < LL; ++l) {
    cvt_swz(Wslice + (size_t)l * CC * SS, WslB + (size_t)l * CC * SS, CC, SS, 1.f);
    cvt_swz(Wfront + (size_t)l * CC * SS, WfrB + (size_t)l * CC * SS, CC, SS, BETA_);
    cvt_swz(Wff1 + (size_t)l * CC * FF, Wf1B + (size_t)l * CC * FF, CC, FF, 1.f);
    cvt_swz(Wff2 + (size_t)l * FF * CC, Wf2B + (size_t)l * FF * CC, FF, CC, 1.f);
  }

  k_embed<<<dim3(BB * NN), dim3(128), 0, stream>>>(inputs, coords, t_norm, freq, embed_W, embed_b, x);

  int nb128 = (NN + 127) / 128;
  int nb32  = (NN + 31) / 32;
  long long nBNC = (long long)BB * NN * CC;

  for (int l = 0; l < LL; ++l) {
    k_ln<<<dim3(BB * NN), dim3(128), 0, stream>>>(x, ln1_g + l * CC, ln1_b + l * CC, h, hbf);
    k_zero<<<dim3((unsigned)((nBNC + 255) / 256)), dim3(256), 0, stream>>>(hloc, nBNC);
    k_zero<<<dim3(1), dim3(64), 0, stream>>>(wsum, (long long)BB * SS);
    k_zero<<<dim3((BB * SS * CC + 255) / 256), dim3(256), 0, stream>>>(slices, (long long)BB * SS * CC);
    long long spmmT = (long long)BB * EE * 4;
    k_spmm<<<dim3((unsigned)((spmmT + 255) / 256)), dim3(256), 0, stream>>>(
        adj_idx, adj_idx + EE, adj_val, h, hloc);
    k_subcvt<<<dim3((unsigned)((nBNC + 255) / 256)), dim3(256), 0, stream>>>(h, hloc, hfbf, nBNC);
    float* w_out = outp + (size_t)BB * NN * CIN + (size_t)l * BB * NN * SS;
    k_logits<<<dim3(nb128, BB), dim3(256), 0, stream>>>(
        hbf, hfbf, WslB + (size_t)l * CC * SS, WfrB + (size_t)l * CC * SS,
        bslice + l * SS, bfront + l * SS, w_out, wbf, wsum, slices);
    k_mha<<<dim3(BB), dim3(256), 0, stream>>>(
        slices, wsum, Win + (size_t)l * CC * 3 * CC, b_in + (size_t)l * 3 * CC,
        Wout + (size_t)l * CC * CC, bout + (size_t)l * CC, sobf);
    k_xupd<<<dim3(nb128, BB), dim3(256), 0, stream>>>(wbf, sobf, x);
    k_ln<<<dim3(BB * NN), dim3(128), 0, stream>>>(x, ln2_g + l * CC, ln2_b + l * CC, h, hfbf);
    k_ffn<<<dim3(nb32, BB), dim3(256), 0, stream>>>(
        hfbf, Wf1B + (size_t)l * CC * FF, bff1 + (size_t)l * FF,
        Wf2B + (size_t)l * FF * CC, bff2 + (size_t)l * CC, x);
  }
  k_proj<<<dim3((BB * NN + 255) / 256), dim3(256), 0, stream>>>(x, proj_W, proj_b, outp);
}
